// RAWTextureDetector_11897059410106
// MI455X (gfx1250) — matrix operand order repaired and checked by reading, not compiled, not run on hardware
//
#include <hip/hip_runtime.h>
#include <hip/hip_bf16.h>
#include <math.h>

typedef __attribute__((ext_vector_type(16))) _Float16 v16h;
typedef __attribute__((ext_vector_type(8)))  float    v8f;

union HFrag { v16h v; _Float16 e[16]; };

#define IMG_W 1024
#define IMG_H 1024
#define HW ((size_t)IMG_W * (size_t)IMG_H)

__device__ __forceinline__ int reflect1024(int i) {
    // jnp.pad mode='reflect' (no edge repeat); pad <= 24 so one reflection suffices
    if (i < 0) return -i;
    if (i > 1023) return 2046 - i;
    return i;
}

// ---------------------------------------------------------------------------
// Kernel 1: fused conv1 (4->32, VALU, leaky 0.2) + conv2 (32->16, WMMA, leaky)
// One block = 16x16 output tile. f1 kept entirely in LDS as f16.
// ---------------------------------------------------------------------------
__global__ __launch_bounds__(256) void conv12_kernel(
    const float* __restrict__ x, const float* __restrict__ w1, const float* __restrict__ b1,
    const float* __restrict__ w2, const float* __restrict__ b2, float* __restrict__ f2) {
    __shared__ float    sX[20 * 20 * 4];        // padded x tile (SAME zero pad)
    __shared__ _Float16 sF1[18 * 18 * 32];      // conv1 output tile (f16)
    __shared__ float    sW1[32 * 36];
    __shared__ _Float16 sW2[16 * 9 * 32];       // [out][tap][in]
    __shared__ float    sB1[32];
    __shared__ float    sB2[16];

    const int tid = threadIdx.x;
    const int bx = blockIdx.x, by = blockIdx.y;

    // --- cooperative loads ---
    for (int i = tid; i < 32 * 36; i += 256) sW1[i] = w1[i];   // natural [o][ci][ky][kx]
    for (int i = tid; i < 16 * 9 * 32; i += 256) {
        const int o = i / 288, r = i % 288;
        const int tap = r / 32, ci = r % 32;
        sW2[o * 288 + tap * 32 + ci] = (_Float16)w2[o * 288 + ci * 9 + tap];
    }
    if (tid < 32) sB1[tid] = b1[tid];
    if (tid < 16) sB2[tid] = b2[tid];
    for (int i = tid; i < 20 * 20 * 4; i += 256) {
        const int p = i >> 2, ci = i & 3;
        const int yy = p / 20, xx = p % 20;
        const int gy = by * 16 - 2 + yy, gx = bx * 16 - 2 + xx;
        float v = 0.0f;
        if (gy >= 0 && gy < IMG_H && gx >= 0 && gx < IMG_W)
            v = x[(size_t)ci * HW + (size_t)gy * IMG_W + gx];
        sX[i] = v;
    }
    __syncthreads();

    // --- conv1 on 18x18 halo tile, 32 out channels ---
    for (int i = tid; i < 18 * 18 * 32; i += 256) {
        const int pix = i >> 5, och = i & 31;
        const int py = pix / 18, px = pix % 18;
        float acc = sB1[och];
        #pragma unroll
        for (int ci = 0; ci < 4; ++ci)
            #pragma unroll
            for (int ky = -1; ky <= 1; ++ky)
                #pragma unroll
                for (int kx = -1; kx <= 1; ++kx)
                    acc += sX[((py + 1 + ky) * 20 + (px + 1 + kx)) * 4 + ci] *
                           sW1[och * 36 + ci * 9 + (ky + 1) * 3 + (kx + 1)];
        acc = acc > 0.0f ? acc : 0.2f * acc;    // leaky relu
        sF1[(py * 18 + px) * 32 + och] = (_Float16)acc;
    }
    __syncthreads();

    // --- conv2 via WMMA f32_16x16x32_f16: M=16 pixels, N=16 out ch, K=9 taps x 32 ---
    const int lane = tid & 31;
    const int wave = tid >> 5;
    const int g    = lane >> 4;     // A K-half selector
    const int mrow = lane & 15;     // A M row
    const int nCol = lane & 15;     // B/C column
    const int kg   = lane >> 4;     // B K-group

    for (int t = wave * 2; t < wave * 2 + 2; ++t) {
        v8f acc = {};
        const int p = t * 16 + mrow;
        const int aly = p >> 4, alx = p & 15;
        #pragma unroll
        for (int tap = 0; tap < 9; ++tap) {
            const int ky = tap / 3 - 1, kx = tap % 3 - 1;
            HFrag A, B;
            const int pixBase = ((aly + 1 + ky) * 18 + (alx + 1 + kx)) * 32;
            #pragma unroll
            for (int j = 0; j < 8; ++j) {
                const int kb = ((j >> 2) << 4) + (g << 3) + ((j & 3) << 1);   // ISA 16b A layout
                A.e[2 * j]     = sF1[pixBase + kb];
                A.e[2 * j + 1] = sF1[pixBase + kb + 1];
                const int kbB = ((j >> 2) << 4) + (kg << 3) + ((j & 3) << 1); // ISA 16b B layout
                B.e[2 * j]     = sW2[nCol * 288 + tap * 32 + kbB];
                B.e[2 * j + 1] = sW2[nCol * 288 + tap * 32 + kbB + 1];
            }
            acc = __builtin_amdgcn_wmma_f32_16x16x32_f16(
                false, A.v, false, B.v, (short)0, acc, false, false);
        }
        #pragma unroll
        for (int vv = 0; vv < 8; ++vv) {
            const int m = (lane >> 4) * 8 + vv;       // ISA C/D layout
            const int pp = t * 16 + m;
            const int ly = pp >> 4, lx = pp & 15;
            const int gy = by * 16 + ly, gx = bx * 16 + lx;
            float val = acc[vv] + sB2[nCol];
            val = val > 0.0f ? val : 0.2f * val;      // leaky relu
            f2[(size_t)nCol * HW + (size_t)gy * IMG_W + gx] = val;
        }
    }
}

// ---------------------------------------------------------------------------
// Kernel 2: conv3 (16->8, no relu) via WMMA, K = 9 taps x 16 = 144 padded to 160
// ---------------------------------------------------------------------------
__global__ __launch_bounds__(256) void conv3_kernel(
    const float* __restrict__ f2, const float* __restrict__ w3,
    const float* __restrict__ b3, float* __restrict__ f3) {
    __shared__ _Float16 sF2[18 * 18 * 16];
    __shared__ _Float16 sW3[16 * 160];     // N padded 8->16, K padded 144->160
    __shared__ float    sB3[8];

    const int tid = threadIdx.x;
    const int bx = blockIdx.x, by = blockIdx.y;

    for (int i = tid; i < 18 * 18 * 16; i += 256) {
        const int p = i >> 4, ci = i & 15;
        const int yy = p / 18, xx = p % 18;
        const int gy = by * 16 - 1 + yy, gx = bx * 16 - 1 + xx;
        float v = 0.0f;
        if (gy >= 0 && gy < IMG_H && gx >= 0 && gx < IMG_W)
            v = f2[(size_t)ci * HW + (size_t)gy * IMG_W + gx];
        sF2[i] = (_Float16)v;
    }
    for (int i = tid; i < 16 * 160; i += 256) sW3[i] = (_Float16)0.0f;
    if (tid < 8) sB3[tid] = b3[tid];
    __syncthreads();
    for (int i = tid; i < 8 * 144; i += 256) {
        const int o = i / 144, r = i % 144;
        const int tap = r >> 4, ci = r & 15;
        sW3[o * 160 + r] = (_Float16)w3[o * 144 + ci * 9 + tap];   // K index = tap*16+ci
    }
    __syncthreads();

    const int lane = tid & 31;
    const int wave = tid >> 5;
    const int g    = lane >> 4;
    const int mrow = lane & 15;
    const int nCol = lane & 15;
    const int kg   = lane >> 4;

    for (int t = wave * 2; t < wave * 2 + 2; ++t) {
        v8f acc = {};
        const int p = t * 16 + mrow;
        const int aly = p >> 4, alx = p & 15;
        #pragma unroll
        for (int kc = 0; kc < 5; ++kc) {
            HFrag A, B;
            #pragma unroll
            for (int j = 0; j < 8; ++j) {
                const int koff = ((j >> 2) << 4) + (g << 3) + ((j & 3) << 1);
                #pragma unroll
                for (int h = 0; h < 2; ++h) {
                    const int Kp = kc * 32 + koff + h;
                    _Float16 av = (_Float16)0.0f;
                    if (Kp < 144) {
                        const int tap = Kp >> 4, ci = Kp & 15;
                        const int ky = tap / 3 - 1, kx = tap % 3 - 1;
                        av = sF2[((aly + 1 + ky) * 18 + (alx + 1 + kx)) * 16 + ci];
                    }
                    A.e[2 * j + h] = av;
                }
                const int KpB = kc * 32 + ((j >> 2) << 4) + (kg << 3) + ((j & 3) << 1);
                B.e[2 * j]     = sW3[nCol * 160 + KpB];      // zero-padded, always safe
                B.e[2 * j + 1] = sW3[nCol * 160 + KpB + 1];
            }
            acc = __builtin_amdgcn_wmma_f32_16x16x32_f16(
                false, A.v, false, B.v, (short)0, acc, false, false);
        }
        if (nCol < 8) {
            #pragma unroll
            for (int vv = 0; vv < 8; ++vv) {
                const int m = (lane >> 4) * 8 + vv;
                const int pp = t * 16 + m;
                const int ly = pp >> 4, lx = pp & 15;
                const int gy = by * 16 + ly, gx = bx * 16 + lx;
                f3[(size_t)nCol * HW + (size_t)gy * IMG_W + gx] = acc[vv] + sB3[nCol];
            }
        }
    }
}

// ---------------------------------------------------------------------------
// Separable box filter: horizontal pass (sums of f and f^2, reflect pad)
// ---------------------------------------------------------------------------
__global__ __launch_bounds__(256) void hpass_kernel(
    const float* __restrict__ f3, float* __restrict__ hs1, float* __restrict__ hs2, int ws) {
    const int y = blockIdx.x, c = blockIdx.y;
    const int r = ws >> 1;
    const float* row = f3 + (size_t)c * HW + (size_t)y * IMG_W;
    float* o1 = hs1 + (size_t)c * HW + (size_t)y * IMG_W;
    float* o2 = hs2 + (size_t)c * HW + (size_t)y * IMG_W;
    for (int xi = threadIdx.x; xi < IMG_W; xi += 256) {
        __builtin_prefetch(row + ((xi + 256 < IMG_W) ? xi + 256 : xi), 0, 0);
        float s1 = 0.0f, s2 = 0.0f;
        for (int d = -r; d <= r; ++d) {
            const float v = row[reflect1024(xi + d)];
            s1 += v; s2 += v * v;
        }
        o1[xi] = s1; o2[xi] = s2;
    }
}

// ---------------------------------------------------------------------------
// Vertical pass + variance + channel-weighted sqrt sum + ^0.8
// ---------------------------------------------------------------------------
__global__ __launch_bounds__(256) void vpass_kernel(
    const float* __restrict__ hs1, const float* __restrict__ hs2,
    float* __restrict__ map, int ws) {
    const size_t idx = (size_t)blockIdx.x * 256 + threadIdx.x;
    const int y = (int)(idx >> 10), xx = (int)(idx & 1023);
    const int r = ws >> 1;
    const float inv = 1.0f / ((float)ws * (float)ws);
    float acc = 0.0f;
    for (int c = 0; c < 8; ++c) {
        float m1 = 0.0f, m2 = 0.0f;
        const float* p1 = hs1 + (size_t)c * HW + xx;
        const float* p2 = hs2 + (size_t)c * HW + xx;
        for (int d = -r; d <= r; ++d) {
            const size_t yoff = (size_t)reflect1024(y + d) * IMG_W;
            m1 += p1[yoff]; m2 += p2[yoff];
        }
        const float mean = m1 * inv;
        float var = m2 * inv - mean * mean;
        var = var < 1e-6f ? 1e-6f : var;
        acc += sqrtf(var) * 0.125f;      // softmax(ones(8)) = 1/8
    }
    map[idx] = powf(acc, 0.8f);
}

// ---------------------------------------------------------------------------
// 1x1 fusion conv over 3 maps + relu
// ---------------------------------------------------------------------------
__global__ __launch_bounds__(256) void fusion_kernel(
    const float* __restrict__ maps, const float* __restrict__ fw,
    const float* __restrict__ fb, float* __restrict__ fused) {
    const size_t i = (size_t)blockIdx.x * 256 + threadIdx.x;
    float v = maps[i] * fw[0] + maps[HW + i] * fw[1] + maps[2 * HW + i] * fw[2] + fb[0];
    fused[i] = v > 0.0f ? v : 0.0f;
}

// ---------------------------------------------------------------------------
// Per-channel sum/sumsq of x (for std with ddof=1)
// ---------------------------------------------------------------------------
__global__ __launch_bounds__(256) void chstats_kernel(
    const float* __restrict__ x, float* __restrict__ chSum, float* __restrict__ chSq) {
    __shared__ float ss[256], sq[256];
    const int c = blockIdx.y, b = blockIdx.x;
    const float* base = x + (size_t)c * HW + (size_t)b * 16384;
    float s = 0.0f, q = 0.0f;
    for (int i = threadIdx.x; i < 16384; i += 256) { const float v = base[i]; s += v; q += v * v; }
    ss[threadIdx.x] = s; sq[threadIdx.x] = q; __syncthreads();
    for (int st = 128; st > 0; st >>= 1) {
        if (threadIdx.x < st) { ss[threadIdx.x] += ss[threadIdx.x + st]; sq[threadIdx.x] += sq[threadIdx.x + st]; }
        __syncthreads();
    }
    if (threadIdx.x == 0) { chSum[c * 64 + b] = ss[0]; chSq[c * 64 + b] = sq[0]; }
}

__global__ __launch_bounds__(256) void minmax_kernel(
    const float* __restrict__ fused, float* __restrict__ pmin, float* __restrict__ pmax) {
    __shared__ float smn[256], smx[256];
    const float* base = fused + (size_t)blockIdx.x * 4096;
    float mn = 3.4e38f, mx = -3.4e38f;
    for (int i = threadIdx.x; i < 4096; i += 256) { const float v = base[i]; mn = fminf(mn, v); mx = fmaxf(mx, v); }
    smn[threadIdx.x] = mn; smx[threadIdx.x] = mx; __syncthreads();
    for (int st = 128; st > 0; st >>= 1) {
        if (threadIdx.x < st) { smn[threadIdx.x] = fminf(smn[threadIdx.x], smn[threadIdx.x + st]);
                                smx[threadIdx.x] = fmaxf(smx[threadIdx.x], smx[threadIdx.x + st]); }
        __syncthreads();
    }
    if (threadIdx.x == 0) { pmin[blockIdx.x] = smn[0]; pmax[blockIdx.x] = smx[0]; }
}

__global__ __launch_bounds__(256) void finalize1_kernel(
    const float* __restrict__ pmin, const float* __restrict__ pmax,
    float* __restrict__ scalars, int* __restrict__ hist) {
    __shared__ float smn[256], smx[256];
    smn[threadIdx.x] = pmin[threadIdx.x]; smx[threadIdx.x] = pmax[threadIdx.x]; __syncthreads();
    for (int st = 128; st > 0; st >>= 1) {
        if (threadIdx.x < st) { smn[threadIdx.x] = fminf(smn[threadIdx.x], smn[threadIdx.x + st]);
                                smx[threadIdx.x] = fmaxf(smx[threadIdx.x], smx[threadIdx.x + st]); }
        __syncthreads();
    }
    if (threadIdx.x == 0) { scalars[0] = smn[0]; scalars[1] = smx[0]; }
    for (int i = threadIdx.x; i < 4096; i += 256) hist[i] = 0;   // fresh histogram every call
}

__global__ __launch_bounds__(256) void hist_kernel(
    const float* __restrict__ fused, const float* __restrict__ scalars, int* __restrict__ hist) {
    __shared__ int lh[4096];
    for (int i = threadIdx.x; i < 4096; i += 256) lh[i] = 0;
    __syncthreads();
    const float mn = scalars[0];
    const float scale = 4096.0f / fmaxf(scalars[1] - mn, 1e-12f);
    const float* base = fused + (size_t)blockIdx.x * 1024;
    for (int i = threadIdx.x; i < 1024; i += 256) {
        int b = (int)((base[i] - mn) * scale);
        b = b < 0 ? 0 : (b > 4095 ? 4095 : b);
        atomicAdd(&lh[b], 1);
    }
    __syncthreads();
    for (int i = threadIdx.x; i < 4096; i += 256) if (lh[i]) atomicAdd(&hist[i], lh[i]);
}

// ---------------------------------------------------------------------------
// Scalar threshold math (quantiles via interpolated histogram CDF)
// ---------------------------------------------------------------------------
__global__ void thresh_kernel(
    const int* __restrict__ hist, float* __restrict__ scalars,
    const float* __restrict__ chSum, const float* __restrict__ chSq,
    const float* __restrict__ cwIn) {
    if (threadIdx.x != 0) return;
    const float mn = scalars[0], mx = scalars[1];
    const float binw = fmaxf(mx - mn, 1e-12f) / 4096.0f;
    const float Nf = (float)HW;
    const float targets[4] = {0.25f, 0.30f, 0.70f, 0.75f};
    float rank[4], q[4];
    for (int i = 0; i < 4; ++i) rank[i] = targets[i] * (Nf - 1.0f);
    float cum = 0.0f; int ti = 0;
    for (int b = 0; b < 4096 && ti < 4; ++b) {
        const float c = (float)hist[b];
        while (ti < 4 && cum + c > rank[ti]) {
            const float frac = c > 0.0f ? (rank[ti] - cum) / c : 0.0f;
            q[ti] = mn + ((float)b + frac) * binw;
            ++ti;
        }
        cum += c;
    }
    for (; ti < 4; ++ti) q[ti] = mx;
    const float q25 = q[0], q30 = q[1], q70 = q[2], q75 = q[3];

    // softmax of channel_weights
    float w[4], wmax = -3.4e38f;
    for (int c = 0; c < 4; ++c) { w[c] = cwIn[c]; wmax = fmaxf(wmax, w[c]); }
    float wsum = 0.0f;
    for (int c = 0; c < 4; ++c) { w[c] = expf(w[c] - wmax); wsum += w[c]; }
    for (int c = 0; c < 4; ++c) w[c] /= wsum;

    float gstd[4], cstd[4];
    for (int c = 0; c < 4; ++c) {
        float s = 0.0f, sq = 0.0f;
        for (int b = 0; b < 64; ++b) { s += chSum[c * 64 + b]; sq += chSq[c * 64 + b]; }
        float var = (sq - s * s / Nf) / (Nf - 1.0f);
        gstd[c] = sqrtf(fmaxf(var, 0.0f));
        cstd[c] = gstd[c] * w[c];         // std(k*x) = k*std(x), k>0
    }
    const float iqr75 = q75 - q25;
    const float lbase = q25 - 0.5f * iqr75;
    const float ubase = q75 + 0.5f * iqr75;
    float lmean = 0.0f, umean = 0.0f;
    for (int c = 0; c < 4; ++c) {
        const float gf = fminf(fmaxf(gstd[c] * 5.0f, 0.5f), 2.0f);
        const float cf = fminf(fmaxf(cstd[c] * 2.0f, 0.8f), 1.2f);
        lmean += lbase * gf * cf; umean += ubase * gf * cf;
    }
    lmean *= 0.25f; umean *= 0.25f;
    float l_t = 0.5f * lmean + 0.5f * q30;
    float u_t = 0.5f * umean + 0.5f * q70;
    const float iqr = q70 - q30;
    if (iqr < 0.01f) { l_t -= iqr; u_t += iqr; }
    const float diff = u_t - l_t;
    const float mid0 = 0.5f * (l_t + u_t);
    if (diff < 0.02f) { l_t = mid0 - 0.01f; u_t = mid0 + 0.01f; }
    scalars[6] = 0.5f * (l_t + u_t);   // midpoint
    scalars[7] = u_t - l_t;            // scale
}

__global__ __launch_bounds__(256) void norm_kernel(
    const float* __restrict__ fused, const float* __restrict__ scalars,
    float* __restrict__ normalized, float* __restrict__ npart) {
    __shared__ float ss[256];
    const float mid = scalars[6], sc = scalars[7];
    const size_t base = (size_t)blockIdx.x * 4096;
    float s = 0.0f;
    for (int i = threadIdx.x; i < 4096; i += 256) {
        const float t = 4.0f * (fused[base + i] - mid) / sc;
        const float n = 1.0f / (1.0f + expf(-t));
        normalized[base + i] = n; s += n;
    }
    ss[threadIdx.x] = s; __syncthreads();
    for (int st = 128; st > 0; st >>= 1) {
        if (threadIdx.x < st) ss[threadIdx.x] += ss[threadIdx.x + st];
        __syncthreads();
    }
    if (threadIdx.x == 0) npart[blockIdx.x] = ss[0];
}

__global__ __launch_bounds__(256) void finalize2_kernel(
    const float* __restrict__ npart, float* __restrict__ scalars) {
    __shared__ float ss[256];
    ss[threadIdx.x] = npart[threadIdx.x]; __syncthreads();
    for (int st = 128; st > 0; st >>= 1) {
        if (threadIdx.x < st) ss[threadIdx.x] += ss[threadIdx.x + st];
        __syncthreads();
    }
    if (threadIdx.x == 0) {
        const float nm = ss[0] / (float)HW;
        float a, b;
        if (nm > 0.65f)      { a = 0.10f; b = 0.50f; }
        else if (nm < 0.35f) { a = 0.20f; b = 0.50f; }
        else                 { a = 0.15f; b = 0.55f; }
        scalars[8] = a; scalars[9] = b;
    }
}

__global__ __launch_bounds__(256) void mask_kernel(
    const float* __restrict__ normalized, const float* __restrict__ scalars,
    float* __restrict__ out) {
    const size_t i = (size_t)blockIdx.x * 256 + threadIdx.x;
    out[i] = scalars[8] + scalars[9] * normalized[i];
}

// ---------------------------------------------------------------------------
extern "C" void kernel_launch(void* const* d_in, const int* in_sizes, int n_in,
                              void* d_out, int out_size, void* d_ws, size_t ws_size,
                              hipStream_t stream) {
    (void)in_sizes; (void)n_in; (void)out_size; (void)ws_size;
    const float* x   = (const float*)d_in[0];
    const float* w1  = (const float*)d_in[1];
    const float* b1  = (const float*)d_in[2];
    const float* w2  = (const float*)d_in[3];
    const float* b2  = (const float*)d_in[4];
    const float* w3  = (const float*)d_in[5];
    const float* b3  = (const float*)d_in[6];
    const float* cw  = (const float*)d_in[7];
    const float* fw  = (const float*)d_in[8];
    const float* fb  = (const float*)d_in[9];
    float* out = (float*)d_out;

    float* W = (float*)d_ws;
    float* f2        = W;                       // 16*HW
    float* f3        = f2 + 16 * HW;            // 8*HW
    float* hs1       = f3 + 8 * HW;             // 8*HW (reused per window size)
    float* hs2       = hs1 + 8 * HW;            // 8*HW
    float* maps      = hs2 + 8 * HW;            // 3*HW
    float* fused     = maps + 3 * HW;           // HW
    float* normalized= fused + HW;              // HW
    float* partMin   = normalized + HW;         // 256
    float* partMax   = partMin + 256;           // 256
    float* chSum     = partMax + 256;           // 4*64
    float* chSq      = chSum + 256;             // 4*64
    float* normPart  = chSq + 256;              // 256
    float* scalars   = normPart + 256;          // 64
    int*   hist      = (int*)(scalars + 64);    // 4096

    conv12_kernel<<<dim3(64, 64), 256, 0, stream>>>(x, w1, b1, w2, b2, f2);
    conv3_kernel <<<dim3(64, 64), 256, 0, stream>>>(f2, w3, b3, f3);

    const int wss[3] = {11, 25, 49};
    for (int s = 0; s < 3; ++s) {
        hpass_kernel<<<dim3(1024, 8), 256, 0, stream>>>(f3, hs1, hs2, wss[s]);
        vpass_kernel<<<4096, 256, 0, stream>>>(hs1, hs2, maps + (size_t)s * HW, wss[s]);
    }
    fusion_kernel<<<4096, 256, 0, stream>>>(maps, fw, fb, fused);

    chstats_kernel<<<dim3(64, 4), 256, 0, stream>>>(x, chSum, chSq);
    minmax_kernel<<<256, 256, 0, stream>>>(fused, partMin, partMax);
    finalize1_kernel<<<1, 256, 0, stream>>>(partMin, partMax, scalars, hist);
    hist_kernel<<<1024, 256, 0, stream>>>(fused, scalars, hist);
    thresh_kernel<<<1, 32, 0, stream>>>(hist, scalars, chSum, chSq, cw);
    norm_kernel<<<256, 256, 0, stream>>>(fused, scalars, normalized, normPart);
    finalize2_kernel<<<1, 256, 0, stream>>>(normPart, scalars);
    mask_kernel<<<4096, 256, 0, stream>>>(normalized, scalars, out);
}
